// MultiHeadSelfAttention_17394617549008
// MI455X (gfx1250) — matrix operand order repaired and checked by reading, not compiled, not run on hardware
//
#include <hip/hip_runtime.h>
#include <hip/hip_bf16.h>

// MHSA forward, MI455X (gfx1250), wave32 + v_wmma_f32_16x16x32_bf16.
// B=8, S=1024, H=1024, NH=16, HD=64.
//
// Roofline: probs store (512 MB) dominates -> mark output streams NT so the
// 192 MB L2 stays dedicated to the reused operands (hs/W f32, q/k/v bf16).

#define B_   8
#define S_   1024
#define H_   1024
#define NH_  16
#define HD_  64

typedef __attribute__((ext_vector_type(16))) __bf16 v16bf;
typedef __attribute__((ext_vector_type(8)))  __bf16 v8bf;
typedef __attribute__((ext_vector_type(8)))  float  v8f;
typedef __attribute__((ext_vector_type(4)))  float  v4f;

static __device__ __forceinline__ v8f wmma_bf(v16bf a, v16bf b, v8f c) {
  // (neg_a, A, neg_b, B, c_mod, C, reuse_a, reuse_b)
  return __builtin_amdgcn_wmma_f32_16x16x32_bf16(false, a, false, b, (short)0, c,
                                                 false, false);
}

static __device__ __forceinline__ v8f vzero8() {
  v8f r;
#pragma unroll
  for (int j = 0; j < 8; ++j) r[j] = 0.0f;
  return r;
}

// A-fragment style: two contiguous 8-element bf16 runs (lane's K-runs).
static __device__ __forceinline__ v16bf ld16_pair(const __bf16* p0, const __bf16* p1) {
  v8bf x = *(const v8bf*)p0;
  v8bf y = *(const v8bf*)p1;
  v16bf r;
#pragma unroll
  for (int j = 0; j < 8; ++j) { r[j] = x[j]; r[8 + j] = y[j]; }
  return r;
}

// B-fragment style: one contiguous 16-element bf16 run.
static __device__ __forceinline__ v16bf ld16_run(const __bf16* p) {
  v8bf x = *(const v8bf*)p;
  v8bf y = *(const v8bf*)(p + 8);
  v16bf r;
#pragma unroll
  for (int j = 0; j < 8; ++j) { r[j] = x[j]; r[8 + j] = y[j]; }
  return r;
}

// Build bf16 fragment from two contiguous 8-element f32 runs (convert inline).
static __device__ __forceinline__ v16bf cvt16_pair(const float* p0, const float* p1) {
  v16bf r;
#pragma unroll
  for (int j = 0; j < 8; ++j) { r[j] = (__bf16)p0[j]; r[8 + j] = (__bf16)p1[j]; }
  return r;
}

static __device__ __forceinline__ v16bf cvt16_run(const float* p) {
  v16bf r;
#pragma unroll
  for (int j = 0; j < 16; ++j) r[j] = (__bf16)p[j];
  return r;
}

// ---------------------------------------------------------------------------
// Kernel 1: QKV projection. out[b,head,s,hd] = (X @ W^T + bias), bf16 output.
// X: [8192,1024] f32 (row-major), W: [1024,1024] f32 (row-major, contract
// along W's row -> B-fragment K-runs are contiguous). Grid: (64, 16), 256 thr.
// Each wave: one 16(M) x 64(N) tile = 4 accumulators; K-loop in steps of 32.
// X and W are L2-resident (192 MB L2), so cross-tile re-reads are cheap.
// ---------------------------------------------------------------------------
__global__ __launch_bounds__(256) void qkv_proj_kernel(
    const float* __restrict__ X, const float* __restrict__ W,
    const float* __restrict__ bias, __bf16* __restrict__ out) {
  const int w    = threadIdx.x >> 5;
  const int lane = threadIdx.x & 31;
  const int lrow = lane & 15;
  const int hi   = (lane >> 4) & 1;   // upper half-wave

  const int m0 = blockIdx.x * 128 + w * 16;   // output row base (this wave)
  const int n0 = blockIdx.y * 64;             // output col base (this wave)

  const float* xrow = X + (size_t)(m0 + lrow) * H_;

  v8f acc[4];
#pragma unroll
  for (int t = 0; t < 4; ++t) acc[t] = vzero8();

  for (int k0 = 0; k0 < H_; k0 += 32) {
    const int lo = hi ? 16 : 0;
    v16bf a = cvt16_pair(xrow + k0 + lo, xrow + k0 + lo + 8);
#pragma unroll
    for (int t = 0; t < 4; ++t) {
      const float* wrow =
          W + (size_t)(n0 + t * 16 + lrow) * H_ + k0 + (hi ? 16 : 0);
      v16bf bfr = cvt16_run(wrow);
      acc[t] = wmma_bf(a, bfr, acc[t]);
    }
  }

  const int bidx = m0 / S_;
  const int s0   = m0 % S_;
#pragma unroll
  for (int t = 0; t < 4; ++t) {
    const int n    = n0 + t * 16 + lrow;   // C/D layout: lane -> N
    const int head = n >> 6;
    const int hd   = n & 63;
    const float bv = bias[n];
    __bf16* ob = out + (((size_t)bidx * NH_ + head) * S_) * HD_ + hd;
#pragma unroll
    for (int r = 0; r < 8; ++r) {          // C/D layout: vgpr r -> M (+8 hi)
      const int s = s0 + r + (hi ? 8 : 0);
      ob[(size_t)s * HD_] = (__bf16)(acc[t][r] + bv);
    }
  }
}

// ---------------------------------------------------------------------------
// Kernel 2: fused attention per (b, h, 64-row q tile).
// Entire 64x1024 f32 score strip lives in LDS (CDNA5: 320 KB/WGP).
//   phase 1: S = (Q K^T)/8 + mask  -> LDS        (WMMA bf16)
//   phase 2: rowmax/exp/rowsum in LDS; write normalized probs (f32, once,
//            non-temporal b128 streaming stores)
//   phase 3: ctx = P V (P re-read from LDS, cvt f32->bf16; V staged
//            transposed in LDS); fold 1/rowsum into NT epilogue stores.
// ---------------------------------------------------------------------------
#define SC_PITCH 1028            // 64x1028 f32, pad avoids 2-way bank conflict
#define VT_PITCH 80              // 64x80 bf16 (160 B rows, 16 B aligned)
#define SC_BYTES (64 * SC_PITCH * 4)
#define VT_BYTES (64 * VT_PITCH * 2)
#define ATTN_LDS (SC_BYTES + VT_BYTES + 640 * 4)

__global__ __launch_bounds__(256) void mhsa_attn_kernel(
    const __bf16* __restrict__ Q, const __bf16* __restrict__ K,
    const __bf16* __restrict__ V, const float* __restrict__ mask,
    float* __restrict__ ctx, float* __restrict__ probs) {
  extern __shared__ char smem[];
  float*  sc     = (float*)smem;                               // [64][1028]
  __bf16* vT     = (__bf16*)(smem + SC_BYTES);                 // [64][80]
  float*  pmax   = (float*)(smem + SC_BYTES + VT_BYTES);       // [64*4]
  float*  psum   = pmax + 256;                                 // [64*4]
  float*  rowmax = pmax + 512;                                 // [64]
  float*  rowinv = pmax + 576;                                 // [64]

  const int qt = blockIdx.x, h = blockIdx.y, b = blockIdx.z;
  const int tid  = threadIdx.x;
  const int w    = tid >> 5;
  const int lane = tid & 31;
  const int lrow = lane & 15;
  const int hi   = (lane >> 4) & 1;

  const size_t bh = (size_t)b * NH_ + h;
  const __bf16* Qh = Q + (bh * S_ + (size_t)qt * 64) * HD_;
  const __bf16* Kh = K + bh * S_ * HD_;
  const __bf16* Vh = V + bh * S_ * HD_;
  const float* mrow = mask + (size_t)b * S_;

  const int m16 = (w & 3) * 16;   // wave's M tile within the 64-row strip

  // ---- phase 1: scores -> LDS ----
  v16bf qa0, qa1;
  {
    const __bf16* qrow = Qh + (size_t)(m16 + lrow) * HD_;
    const int lo = hi ? 16 : 0;
    qa0 = ld16_pair(qrow + lo,      qrow + lo + 8);         // K = 0..31
    qa1 = ld16_pair(qrow + 32 + lo, qrow + 32 + lo + 8);    // K = 32..63
  }
  const int nt0 = (w >> 2) * 32;   // 32 N-tiles per wave (half the keys)
  for (int t = 0; t < 32; ++t) {
    const int ncol = (nt0 + t) * 16 + lrow;  // key position (B lane -> N)
    const __bf16* krow = Kh + (size_t)ncol * HD_ + (hi ? 16 : 0);
    v8f acc = vzero8();
    acc = wmma_bf(qa0, ld16_run(krow), acc);
    acc = wmma_bf(qa1, ld16_run(krow + 32), acc);
    const float mv = mrow[ncol];
#pragma unroll
    for (int r = 0; r < 8; ++r) {
      const int m = m16 + r + (hi ? 8 : 0);
      sc[m * SC_PITCH + ncol] = acc[r] * 0.125f + mv;   // 1/sqrt(64)
    }
  }
  __syncthreads();

  // ---- phase 2: softmax over each of the 64 rows (1024 cols) ----
  {
    const int row = tid >> 2, part = tid & 3;
    const float* srow = sc + row * SC_PITCH + part * 256;
    float mx = -3.4e38f;
    for (int c = 0; c < 256; ++c) mx = fmaxf(mx, srow[c]);
    pmax[row * 4 + part] = mx;
  }
  __syncthreads();
  if (tid < 64) {
    rowmax[tid] = fmaxf(fmaxf(pmax[tid * 4], pmax[tid * 4 + 1]),
                        fmaxf(pmax[tid * 4 + 2], pmax[tid * 4 + 3]));
  }
  __syncthreads();
  {
    const int row = tid >> 2, part = tid & 3;
    float* srow = sc + row * SC_PITCH + part * 256;
    const float mx = rowmax[row];
    float sum = 0.0f;
    for (int c = 0; c < 256; ++c) {
      const float e = __expf(srow[c] - mx);
      srow[c] = e;                 // keep unnormalized exp in LDS
      sum += e;
    }
    psum[row * 4 + part] = sum;
  }
  __syncthreads();
  if (tid < 64) {
    rowinv[tid] = 1.0f / (psum[tid * 4] + psum[tid * 4 + 1] +
                          psum[tid * 4 + 2] + psum[tid * 4 + 3]);
  }
  __syncthreads();

  // normalized probs -> global: non-temporal b128 streaming stores (written
  // once, never re-read on-device -> don't let 512 MB wash through L2)
  {
    float* pb = probs + (bh * S_ + (size_t)qt * 64) * S_;
    for (int idx = tid; idx < 64 * 256; idx += 256) {
      const int r  = idx >> 8;           // row 0..63
      const int c4 = (idx & 255) * 4;    // col (x4)
      const float inv = rowinv[r];
      const float* s4 = sc + r * SC_PITCH + c4;
      v4f v;
      v[0] = s4[0] * inv;
      v[1] = s4[1] * inv;
      v[2] = s4[2] * inv;
      v[3] = s4[3] * inv;
      __builtin_nontemporal_store(v, (v4f*)(pb + (size_t)r * S_ + c4));
    }
  }

  // ---- phase 3: ctx = P V (fold rowinv into epilogue) ----
  v8f cacc[2];
  cacc[0] = vzero8();
  cacc[1] = vzero8();
  const int n16a = (w >> 2) * 2;   // wave's 2 HD tiles (of 4)

  for (int kc = 0; kc < S_; kc += 64) {
    __syncthreads();   // protect vT reuse across iterations
    {
      // stage V[kc..kc+64, 0..64] transposed: vT[hd][k]
      const int kk = tid >> 2, part = tid & 3;
      const __bf16* vp = Vh + (size_t)(kc + kk) * HD_ + part * 16;
      v8bf a = *(const v8bf*)vp;
      v8bf c = *(const v8bf*)(vp + 8);
#pragma unroll
      for (int j = 0; j < 8; ++j) {
        vT[(part * 16 + j) * VT_PITCH + kk]     = a[j];
        vT[(part * 16 + 8 + j) * VT_PITCH + kk] = c[j];
      }
    }
    __syncthreads();
#pragma unroll
    for (int ks = 0; ks < 2; ++ks) {
      // A fragment: P rows m16.., K = kc + ks*32, from LDS f32 -> bf16
      const float* prow =
          sc + (m16 + lrow) * SC_PITCH + kc + ks * 32 + (hi ? 16 : 0);
      v16bf pa = cvt16_pair(prow, prow + 8);
#pragma unroll
      for (int t = 0; t < 2; ++t) {
        const int n = (n16a + t) * 16 + lrow;   // hd column
        v16bf vbf = ld16_run(vT + n * VT_PITCH + ks * 32 + (hi ? 16 : 0));
        cacc[t] = wmma_bf(pa, vbf, cacc[t]);
      }
    }
  }

  {
#pragma unroll
    for (int t = 0; t < 2; ++t) {
      const int n = (n16a + t) * 16 + lrow;
      float* ob = ctx + ((size_t)b * S_ + (size_t)qt * 64) * H_ + h * HD_ + n;
#pragma unroll
      for (int r = 0; r < 8; ++r) {
        const int m = m16 + r + (hi ? 8 : 0);
        __builtin_nontemporal_store(cacc[t][r] * rowinv[m],
                                    ob + (size_t)m * H_);
      }
    }
  }
}

// ---------------------------------------------------------------------------
extern "C" void kernel_launch(void* const* d_in, const int* in_sizes, int n_in,
                              void* d_out, int out_size, void* d_ws,
                              size_t ws_size, hipStream_t stream) {
  (void)in_sizes; (void)n_in; (void)out_size; (void)ws_size;

  const float* hs   = (const float*)d_in[0];
  const float* mask = (const float*)d_in[1];
  const float* Wq   = (const float*)d_in[2];
  const float* bq   = (const float*)d_in[3];
  const float* Wk   = (const float*)d_in[4];
  const float* bk   = (const float*)d_in[5];
  const float* Wv   = (const float*)d_in[6];
  const float* bv   = (const float*)d_in[7];

  float* ctx   = (float*)d_out;                       // [B,S,H] f32
  float* probs = ctx + (size_t)B_ * S_ * H_;          // [B,NH,S,S] f32

  // workspace: q/k/v in bf16, [B,NH,S,HD] each (16 MB x 3 = 48 MB)
  __bf16* qb = (__bf16*)d_ws;
  __bf16* kb = qb + (size_t)B_ * NH_ * S_ * HD_;
  __bf16* vb = kb + (size_t)B_ * NH_ * S_ * HD_;

  const dim3 pg(64, 16, 1);   // M/128 x N/64
  qkv_proj_kernel<<<pg, 256, 0, stream>>>(hs, Wq, bq, qb);
  qkv_proj_kernel<<<pg, 256, 0, stream>>>(hs, Wk, bk, kb);
  qkv_proj_kernel<<<pg, 256, 0, stream>>>(hs, Wv, bv, vb);

  (void)hipFuncSetAttribute((const void*)mhsa_attn_kernel,
                            hipFuncAttributeMaxDynamicSharedMemorySize,
                            ATTN_LDS);
  mhsa_attn_kernel<<<dim3(S_ / 64, NH_, B_), 256, ATTN_LDS, stream>>>(
      qb, kb, vb, mask, ctx, probs);
}
